// FullCausalAttention_25933012533613
// MI455X (gfx1250) — matrix operand order repaired and checked by reading, not compiled, not run on hardware
//
#include <hip/hip_runtime.h>

typedef __attribute__((ext_vector_type(16))) _Float16 v16h;
typedef __attribute__((ext_vector_type(8)))  _Float16 v8h;
typedef __attribute__((ext_vector_type(4)))  _Float16 v4h;
typedef __attribute__((ext_vector_type(8)))  float    v8f;
typedef __attribute__((ext_vector_type(4)))  int      v4i;

union FragA { v16h v; v8h h[2]; };

#define WMMA_F32_F16(a, b, c) \
  __builtin_amdgcn_wmma_f32_16x16x32_f16(false, (a), false, (b), (short)0, (c), false, false)

#define TSEQ 2048

// ---- optional CDNA5 async global->LDS path (compile-safe fallback) --------
#if defined(__has_builtin)
#  if __has_builtin(__builtin_amdgcn_global_load_async_to_lds_b128) && \
      __has_builtin(__builtin_amdgcn_s_wait_asynccnt)
#    define HAVE_ASYNC_LDS 1
#  endif
#endif
#ifndef HAVE_ASYNC_LDS
#  define HAVE_ASYNC_LDS 0
#endif

#if HAVE_ASYNC_LDS
// prototype (from hipcc diagnostics):
//   void __builtin_amdgcn_global_load_async_to_lds_b128(v4i* gsrc,
//        __attribute__((address_space(3))) v4i* ldsdst, imm offset, imm cpol)
typedef __attribute__((address_space(3))) v4i lv4i_t;
#define AS_GLOBAL(p) ((v4i*)(p))
#define AS_LOCAL(p)  ((lv4i_t*)(unsigned long long)(p))
#endif

// ---------------------------------------------------------------------------
// fp32 -> f16 conversion (vectorized; all element counts divisible by 4)
// ---------------------------------------------------------------------------
__global__ void cvt_f32_to_f16(const float4* __restrict__ in,
                               v4h* __restrict__ out, int n4) {
  int i = blockIdx.x * blockDim.x + threadIdx.x;
  if (i < n4) {
    float4 f = in[i];
    v4h o;
    o[0] = (_Float16)f.x; o[1] = (_Float16)f.y;
    o[2] = (_Float16)f.z; o[3] = (_Float16)f.w;
    out[i] = o;
  }
}

// ---------------------------------------------------------------------------
// WMMA GEMM:  C[M,N] = A[M,K] * Bw[N,K]^T   (Bw row-major over K)
// 256 threads = 8 waves; waves arranged 4(M) x 2(N); each wave owns 32x64.
// ---------------------------------------------------------------------------
__launch_bounds__(256)
__global__ void gemm_wmma_f16(const _Float16* __restrict__ A,
                              const _Float16* __restrict__ Bw,
                              float* __restrict__ Cf,
                              _Float16* __restrict__ Ch,
                              int M, int N, int K) {
  const int lane  = threadIdx.x & 31;
  const int wid   = threadIdx.x >> 5;
  const int wm    = wid & 3;
  const int wn    = wid >> 2;
  const int m0    = blockIdx.y * 128 + wm * 32;
  const int n0    = blockIdx.x * 128 + wn * 64;
  const int l16   = lane & 15;
  const int lhalf = lane >> 4;

  v8f acc[2][4];
#pragma unroll
  for (int i = 0; i < 2; ++i)
#pragma unroll
    for (int j = 0; j < 4; ++j) acc[i][j] = (v8f)(0.0f);

  for (int k0 = 0; k0 < K; k0 += 32) {
    FragA a[2];
#pragma unroll
    for (int mi = 0; mi < 2; ++mi) {
      const _Float16* ap = A + (size_t)(m0 + mi * 16 + l16) * K + k0 + lhalf * 16;
      a[mi].h[0] = *(const v8h*)(ap);
      a[mi].h[1] = *(const v8h*)(ap + 8);
    }
    FragA b[4];
#pragma unroll
    for (int ni = 0; ni < 4; ++ni) {
      const _Float16* bp = Bw + (size_t)(n0 + ni * 16 + l16) * K + k0 + lhalf * 16;
      b[ni].h[0] = *(const v8h*)(bp);
      b[ni].h[1] = *(const v8h*)(bp + 8);
    }
#pragma unroll
    for (int mi = 0; mi < 2; ++mi)
#pragma unroll
      for (int ni = 0; ni < 4; ++ni)
        acc[mi][ni] = WMMA_F32_F16(a[mi].v, b[ni].v, acc[mi][ni]);
  }

#pragma unroll
  for (int mi = 0; mi < 2; ++mi)
#pragma unroll
    for (int ni = 0; ni < 4; ++ni)
#pragma unroll
      for (int r = 0; r < 8; ++r) {
        int row = m0 + mi * 16 + r + lhalf * 8;
        int col = n0 + ni * 16 + l16;
        float v = acc[mi][ni][r];
        if (Ch) Ch[(size_t)row * N + col] = (_Float16)v;
        else    Cf[(size_t)row * N + col] = v;
      }
}

// ---------------------------------------------------------------------------
// V transpose: qkv [B*T,3072] (V at col 2048 + h*64 + d)  ->  Vt[(b*16+h)*64+d, t]
// Block handles one (b,h) head x 64-token tile. LDS tile 64x64 (stride 72).
// Uses CDNA5 async global->LDS loads when the builtin is available.
// ---------------------------------------------------------------------------
__launch_bounds__(256)
__global__ void transpose_v(const _Float16* __restrict__ QKV,
                            _Float16* __restrict__ Vt) {
  __shared__ _Float16 tile[64 * 72];
  const int bh = blockIdx.x >> 5;          // b*16 + h   (0..31)
  const int tt = blockIdx.x & 31;          // 64-token tile (0..31)
  const int b  = bh >> 4;
  const int h  = bh & 15;

  const int trow = threadIdx.x >> 2;       // 0..63  (token within tile)
  const int dseg = (threadIdx.x & 3) * 16; // 0,16,32,48

  const _Float16* src =
      QKV + ((size_t)b * TSEQ + tt * 64 + trow) * 3072 + 2048 + h * 64 + dseg;
  _Float16* dst = &tile[trow * 72 + dseg];

#if HAVE_ASYNC_LDS
  __builtin_amdgcn_global_load_async_to_lds_b128(AS_GLOBAL(src),     AS_LOCAL(dst),     0, 0);
  __builtin_amdgcn_global_load_async_to_lds_b128(AS_GLOBAL(src + 8), AS_LOCAL(dst + 8), 0, 0);
  __builtin_amdgcn_s_wait_asynccnt(0);
#else
  *(v8h*)dst       = *(const v8h*)src;
  *(v8h*)(dst + 8) = *(const v8h*)(src + 8);
#endif
  __syncthreads();

  const int drow = threadIdx.x >> 2;       // 0..63  (d)
  const int tseg = (threadIdx.x & 3) * 16; // token run within tile
  union { v8h h[2]; _Float16 e[16]; } o;
#pragma unroll
  for (int i = 0; i < 16; ++i) o.e[i] = tile[(tseg + i) * 72 + drow];

  _Float16* op = Vt + ((size_t)bh * 64 + drow) * TSEQ + tt * 64 + tseg;
  *(v8h*)op       = o.h[0];
  *(v8h*)(op + 8) = o.h[1];
}

// ---------------------------------------------------------------------------
// Flash causal attention. 128 threads = 4 waves, one 16-query tile per wave.
// Q,K from qkv [B*T,3072]; V from transposed Vt [(b*16+h)*64+d, t].
// ---------------------------------------------------------------------------
__launch_bounds__(128)
__global__ void flash_attn_wmma(const _Float16* __restrict__ QKV,
                                const _Float16* __restrict__ Vt,
                                _Float16* __restrict__ Oa) {
  __shared__ _Float16 lds_p[4][16 * 32];

  const int lane  = threadIdx.x & 31;
  const int wid   = threadIdx.x >> 5;
  const int l16   = lane & 15;
  const int lhalf = lane >> 4;

  const int gq    = blockIdx.x * 4 + wid;
  const int qt    = gq & 127;
  const int bh    = gq >> 7;
  const int h     = bh & 15;
  const int b     = bh >> 4;
  const int qbase = qt * 16;

  const size_t rs = 3072;
  const _Float16* Qp  = QKV + (size_t)b * TSEQ * rs + h * 64;
  const _Float16* Kp  = Qp + 1024;
  const _Float16* Vtp = Vt + (size_t)bh * 64 * TSEQ;

  FragA qf[2];
#pragma unroll
  for (int c = 0; c < 2; ++c) {
    const _Float16* qp = Qp + (size_t)(qbase + l16) * rs + c * 32 + lhalf * 16;
    qf[c].h[0] = *(const v8h*)(qp);
    qf[c].h[1] = *(const v8h*)(qp + 8);
  }

  v8f oacc[4];
#pragma unroll
  for (int n = 0; n < 4; ++n) oacc[n] = (v8f)(0.0f);
  float mrow[8], lrow[8];
#pragma unroll
  for (int r = 0; r < 8; ++r) { mrow[r] = -1e30f; lrow[r] = 0.0f; }

  const float scale = 0.125f;
  const int kend = qbase + 16;        // all key indices touched stay < TSEQ

  for (int kb = 0; kb < kend; kb += 32) {
    // ---- S = Q K^T over this 32-key chunk
    v8f s[2];
    s[0] = (v8f)(0.0f);
    s[1] = (v8f)(0.0f);
#pragma unroll
    for (int c = 0; c < 2; ++c) {
#pragma unroll
      for (int t = 0; t < 2; ++t) {
        const _Float16* kp = Kp + (size_t)(kb + t * 16 + l16) * rs + c * 32 + lhalf * 16;
        FragA bk;
        bk.h[0] = *(const v8h*)(kp);
        bk.h[1] = *(const v8h*)(kp + 8);
        s[t] = WMMA_F32_F16(qf[c].v, bk.v, s[t]);
      }
    }

    // ---- online softmax
    float p0a[8], p1a[8];
#pragma unroll
    for (int r = 0; r < 8; ++r) {
      int row  = qbase + r + lhalf * 8;
      int key0 = kb + l16;
      int key1 = key0 + 16;
      float v0 = s[0][r] * scale;
      float v1 = s[1][r] * scale;
      if (key0 > row) v0 = -1e30f;
      if (key1 > row) v1 = -1e30f;
      float mx = fmaxf(v0, v1);
#pragma unroll
      for (int off = 1; off < 16; off <<= 1) mx = fmaxf(mx, __shfl_xor(mx, off));
      float mnew  = fmaxf(mrow[r], mx);
      float alpha = __expf(mrow[r] - mnew);
      mrow[r] = mnew;
      float p0 = __expf(v0 - mnew);
      float p1 = __expf(v1 - mnew);
      float sum = p0 + p1;
#pragma unroll
      for (int off = 1; off < 16; off <<= 1) sum += __shfl_xor(sum, off);
      lrow[r] = lrow[r] * alpha + sum;
#pragma unroll
      for (int n = 0; n < 4; ++n) oacc[n][r] *= alpha;
      p0a[r] = p0;
      p1a[r] = p1;
    }

    // ---- P (C-layout f32) -> LDS -> A-layout f16 fragment
    _Float16* lp = lds_p[wid];
#pragma unroll
    for (int r = 0; r < 8; ++r) {
      int rl = r + lhalf * 8;
      lp[rl * 32 + l16]      = (_Float16)p0a[r];
      lp[rl * 32 + 16 + l16] = (_Float16)p1a[r];
    }
    asm volatile("s_wait_dscnt 0" ::: "memory");
    FragA pf;
    {
      const _Float16* pp = lp + (size_t)l16 * 32 + lhalf * 16;
      pf.h[0] = *(const v8h*)(pp);
      pf.h[1] = *(const v8h*)(pp + 8);
    }

    // ---- O += P V : B-fragment is contiguous along t in Vt (two b128 loads)
#pragma unroll
    for (int n = 0; n < 4; ++n) {
      const _Float16* vp = Vtp + (size_t)(n * 16 + l16) * TSEQ + kb + lhalf * 16;
      FragA vf;
      vf.h[0] = *(const v8h*)(vp);
      vf.h[1] = *(const v8h*)(vp + 8);
      oacc[n] = WMMA_F32_F16(pf.v, vf.v, oacc[n]);
    }
  }

  // ---- normalize and write [B*T, 1024] f16 (head-merged)
#pragma unroll
  for (int n = 0; n < 4; ++n)
#pragma unroll
    for (int r = 0; r < 8; ++r) {
      int row = qbase + r + lhalf * 8;
      int d   = n * 16 + l16;
      float v = oacc[n][r] / lrow[r];
      Oa[((size_t)b * TSEQ + row) * 1024 + h * 64 + d] = (_Float16)v;
    }
}

// ---------------------------------------------------------------------------
extern "C" void kernel_launch(void* const* d_in, const int* in_sizes, int n_in,
                              void* d_out, int out_size, void* d_ws, size_t ws_size,
                              hipStream_t stream) {
  const float* x    = (const float*)d_in[0];   // [2,2048,1024]
  const float* Wqkv = (const float*)d_in[1];   // [3072,1024]
  const float* Wout = (const float*)d_in[2];   // [1024,1024]
  float* out = (float*)d_out;                  // [2,2048,1024]

  const int BT = 2 * TSEQ;   // 4096
  const int D  = 1024;
  const int E3 = 3072;

  _Float16* ws     = (_Float16*)d_ws;
  _Float16* x16    = ws;                          // 4096*1024
  _Float16* wqkv16 = x16 + (size_t)BT * D;        // 3072*1024
  _Float16* wout16 = wqkv16 + (size_t)E3 * D;     // 1024*1024
  _Float16* qkv16  = wout16 + (size_t)D * D;      // 4096*3072
  _Float16* attn16 = qkv16 + (size_t)BT * E3;     // 4096*1024
  _Float16* vt16   = attn16 + (size_t)BT * D;     // 32*64*2048 = 4096*1024

  cvt_f32_to_f16<<<(BT * D / 4 + 255) / 256, 256, 0, stream>>>(
      (const float4*)x, (v4h*)x16, BT * D / 4);
  cvt_f32_to_f16<<<(E3 * D / 4 + 255) / 256, 256, 0, stream>>>(
      (const float4*)Wqkv, (v4h*)wqkv16, E3 * D / 4);
  cvt_f32_to_f16<<<(D * D / 4 + 255) / 256, 256, 0, stream>>>(
      (const float4*)Wout, (v4h*)wout16, D * D / 4);

  // QKV projection: [4096,1024] x [3072,1024]^T -> f16 [4096,3072]
  dim3 g1(E3 / 128, BT / 128);
  gemm_wmma_f16<<<g1, 256, 0, stream>>>(x16, wqkv16, nullptr, qkv16, BT, E3, D);

  // V transpose into [(b*16+h)*64+d, t]
  transpose_v<<<32 * (TSEQ / 64), 256, 0, stream>>>(qkv16, vt16);

  // causal attention -> f16 [4096,1024]; 4096 q-tiles, 4 waves/block
  flash_attn_wmma<<<1024, 128, 0, stream>>>(qkv16, vt16, attn16);

  // output projection: [4096,1024] x [1024,1024]^T -> f32 d_out
  dim3 g2(D / 128, BT / 128);
  gemm_wmma_f16<<<g2, 256, 0, stream>>>(attn16, wout16, out, nullptr, BT, D, D);
}
